// FusedALiBiAttention_61237643706933
// MI455X (gfx1250) — matrix operand order repaired and checked by reading, not compiled, not run on hardware
//
#include <hip/hip_runtime.h>

// ---------------------------------------------------------------------------
// Fused ALiBi causal attention for MI455X (gfx1250, wave32, WMMA bf16).
//   K0: one-shot f32 -> bf16 conversion of x, W_qkv, W_out
//   K1: qkv_bf16 = x_bf16 @ Wqkv_bf16 + b_qkv   (WMMA, f32 accum, bf16 out)
//   K2: flash-style causal ALiBi attention      (WMMA, online softmax, bf16 out)
//   K3: out_f32 = attn_bf16 @ Wout_bf16 + b_out (WMMA, f32 accum, f32 out)
// Hot loops are ds_load_b128 + v_wmma; softmax reductions use ds_swizzle
// (constant XOR butterfly) instead of ds_bpermute; ALiBi bias is hoisted.
// ---------------------------------------------------------------------------

typedef __attribute__((ext_vector_type(4)))  unsigned int v4u;
typedef __attribute__((ext_vector_type(8)))  unsigned int v8u;
typedef __attribute__((ext_vector_type(8)))  float        v8f;
typedef __attribute__((ext_vector_type(16))) __bf16       v16bf;

#define S_LEN 2048
#define NH    16
#define HD    64
#define DM    1024

__device__ __forceinline__ unsigned short f2bf(float f) {
  unsigned int u = __float_as_uint(f);
  unsigned int r = (u + 0x7FFFu + ((u >> 16) & 1u)) >> 16;  // RNE
  return (unsigned short)r;
}
__device__ __forceinline__ unsigned int pk2(float a, float b) {
  return (unsigned int)f2bf(a) | ((unsigned int)f2bf(b) << 16);
}
__device__ __forceinline__ v8f wmma_bf16(v8u a, v8u b, v8f c) {
  return __builtin_amdgcn_wmma_f32_16x16x32_bf16(
      false, __builtin_bit_cast(v16bf, a),
      false, __builtin_bit_cast(v16bf, b),
      (short)0, c, false, false);
}
// 16-element bf16 fragment = two contiguous 8-element (16B) groups.
__device__ __forceinline__ v8u ld_frag(const unsigned short* p, int gap) {
  v4u lo = *(const v4u*)p;
  v4u hi = *(const v4u*)(p + gap);
  return __builtin_shufflevector(lo, hi, 0, 1, 2, 3, 4, 5, 6, 7);
}
// DS_SWIZZLE_B32 group-of-32 XOR butterfly (SWAPXn): offset = xor<<10 | 0x1f.
template <int PAT>
__device__ __forceinline__ float ds_swz(float v) {
  return __int_as_float(__builtin_amdgcn_ds_swizzle(__float_as_int(v), PAT));
}
__device__ __forceinline__ float bfly_max(float v) {
  v = fmaxf(v, ds_swz<0x041f>(v));  // xor 1
  v = fmaxf(v, ds_swz<0x081f>(v));  // xor 2
  v = fmaxf(v, ds_swz<0x101f>(v));  // xor 4
  v = fmaxf(v, ds_swz<0x201f>(v));  // xor 8 (stays within 16-lane half)
  return v;
}
__device__ __forceinline__ float bfly_sum(float v) {
  v += ds_swz<0x041f>(v);
  v += ds_swz<0x081f>(v);
  v += ds_swz<0x101f>(v);
  v += ds_swz<0x201f>(v);
  return v;
}

// ---------------------------------------------------------------------------
// K0: vectorized f32 -> bf16 (8 elements / thread). n multiple of 8.
// ---------------------------------------------------------------------------
__global__ __launch_bounds__(256) void cvt_f32_bf16(
    const float* __restrict__ in, unsigned short* __restrict__ out, int n8) {
  int i = blockIdx.x * 256 + threadIdx.x;
  if (i >= n8) return;
  const float4* p = (const float4*)in + (size_t)i * 2;
  float4 a = p[0], b = p[1];
  v4u o;
  o[0] = pk2(a.x, a.y); o[1] = pk2(a.z, a.w);
  o[2] = pk2(b.x, b.y); o[3] = pk2(b.z, b.w);
  *((v4u*)out + i) = o;
}

// ---------------------------------------------------------------------------
// GEMM: C[MxN] = A[MxK] * B[KxN] + bias[N], A/B bf16, accum f32.
// WG = 256 threads = 8 waves; tile 128x128, K-step 32, double-buffered LDS.
// Wave (m,n) grid 2x4; each wave owns 64x32 = 8 WMMAs per K-step.
// Requires M%128==0, N%128==0, K%32==0.
// ---------------------------------------------------------------------------
#define LDS_S 40       // 32 + 8 pad ushorts; row stride 80B = 5*16B
#define BUF_E (128 * LDS_S)

template <bool OUT_F32>
__global__ __launch_bounds__(256) void gemm_bf16_wmma(
    const unsigned short* __restrict__ A, const unsigned short* __restrict__ B,
    const float* __restrict__ bias, void* __restrict__ Cout,
    int N, int K, int lda, int ldb, int ldc) {
  __shared__ __align__(16) unsigned short sA[2 * BUF_E];   // [m][k]
  __shared__ __align__(16) unsigned short sBt[2 * BUF_E];  // [n][k] transposed

  const int tid  = threadIdx.x;
  const int wid  = tid >> 5;  // wave32
  const int lane = tid & 31;
  const int half = lane >> 4;
  const int l15  = lane & 15;
  const int m0 = blockIdx.y * 128;
  const int n0 = blockIdx.x * 128;
  const int wm = (wid & 1) * 64;
  const int wn = (wid >> 1) * 32;

  v8f acc[4][2];
#pragma unroll
  for (int i = 0; i < 4; ++i)
#pragma unroll
    for (int j = 0; j < 2; ++j)
#pragma unroll
      for (int r = 0; r < 8; ++r) acc[i][j][r] = 0.0f;

  auto stage = [&](int k0, int bsel) {
    unsigned short* ad = sA + bsel * BUF_E;
    unsigned short* bd = sBt + bsel * BUF_E;
    // A: 512 x 16B chunks, 2 per thread, straight b128 copies.
#pragma unroll
    for (int i = 0; i < 2; ++i) {
      int e = tid + i * 256;
      int r = e >> 2, c = (e & 3) * 8;
      *(v4u*)&ad[r * LDS_S + c] =
          *(const v4u*)&A[(size_t)(m0 + r) * lda + k0 + c];
    }
    // B: b32 loads (2 bf16 along n), transposed b16 stores into [n][k].
#pragma unroll
    for (int i = 0; i < 8; ++i) {
      int e = tid + i * 256;
      int r = e >> 6, c2 = (e & 63) * 2;
      unsigned u = *(const unsigned*)&B[(size_t)(k0 + r) * ldb + n0 + c2];
      bd[c2 * LDS_S + r]       = (unsigned short)u;
      bd[(c2 + 1) * LDS_S + r] = (unsigned short)(u >> 16);
    }
  };

  stage(0, 0);
  __syncthreads();
  int buf = 0;
  for (int k0 = 0; k0 < K; k0 += 32) {
    if (k0 + 32 < K) stage(k0 + 32, buf ^ 1);  // prefetch next K-step
    const unsigned short* ar = sA + buf * BUF_E;
    const unsigned short* br = sBt + buf * BUF_E;
    v8u af[4], bfr[2];
#pragma unroll
    for (int i = 0; i < 4; ++i)
      af[i] = ld_frag(&ar[(wm + i * 16 + l15) * LDS_S + half * 8], 16);
#pragma unroll
    for (int j = 0; j < 2; ++j)
      bfr[j] = ld_frag(&br[(wn + j * 16 + l15) * LDS_S + half * 8], 16);
#pragma unroll
    for (int i = 0; i < 4; ++i)
#pragma unroll
      for (int j = 0; j < 2; ++j)
        acc[i][j] = wmma_bf16(af[i], bfr[j], acc[i][j]);
    __syncthreads();
    buf ^= 1;
  }

#pragma unroll
  for (int i = 0; i < 4; ++i) {
#pragma unroll
    for (int j = 0; j < 2; ++j) {
      int col = n0 + wn + j * 16 + l15;
      float bv = bias[col];
#pragma unroll
      for (int r = 0; r < 8; ++r) {
        int row = m0 + wm + i * 16 + half * 8 + r;
        float v = acc[i][j][r] + bv;
        if (OUT_F32)
          ((float*)Cout)[(size_t)row * ldc + col] = v;
        else
          ((unsigned short*)Cout)[(size_t)row * ldc + col] = f2bf(v);
      }
    }
  }
}

// ---------------------------------------------------------------------------
// K2: fused causal ALiBi attention (flash-style, online softmax).
// qkv: bf16 [B*S, 3*DM]; Q at col h*64, K at 1024+h*64, V at 2048+h*64.
// Grid (S/64, NH, B); block 128 = 4 waves; each wave owns 16 queries.
// ---------------------------------------------------------------------------
__global__ __launch_bounds__(128) void attn_alibi_wmma(
    const unsigned short* __restrict__ qkv, unsigned short* __restrict__ out) {
  __shared__ __align__(16) unsigned short sK[32 * 72];     // [key][dim]
  __shared__ __align__(16) unsigned short sVt[64 * LDS_S]; // [dim][key]
  __shared__ __align__(16) unsigned short sP[4 * 16 * 32]; // per-wave P 16x32

  const int tid  = threadIdx.x;
  const int wid  = tid >> 5;
  const int lane = tid & 31;
  const int half = lane >> 4;
  const int l15  = lane & 15;
  const int q0 = blockIdx.x * 64;
  const int h  = blockIdx.y;
  const int b  = blockIdx.z;
  const int qw = q0 + wid * 16;
  const float slope = exp2f(-0.5f * (float)(h + 1));  // ALiBi, NH=16 (pow2)

  // Q fragments (A-layout) straight from bf16 global.
  const unsigned short* qp =
      qkv + ((size_t)(b * S_LEN + qw + l15)) * (3 * DM) + h * HD;
  v8u qf[2];
#pragma unroll
  for (int t = 0; t < 2; ++t)
    qf[t] = ld_frag(qp + t * 32 + half * 8, 16);

  v8f acc[4];
#pragma unroll
  for (int n = 0; n < 4; ++n)
#pragma unroll
    for (int r = 0; r < 8; ++r) acc[n][r] = 0.0f;
  float m_run[8], l_run[8];
  int   qi[8];      // this lane-half's query rows
  float qb[8];      // hoisted slope*q
#pragma unroll
  for (int r = 0; r < 8; ++r) {
    m_run[r] = -__builtin_inff();
    l_run[r] = 0.0f;
    qi[r] = qw + half * 8 + r;
    qb[r] = slope * (float)qi[r];
  }

  const int kmax = q0 + 63;
  for (int kb = 0; kb <= kmax; kb += 32) {
    // Stage K natural [key][dim] with b128 copies (2 per thread).
#pragma unroll
    for (int i = 0; i < 2; ++i) {
      int e = tid + i * 128;
      int key = e >> 3, c = (e & 7) * 8;
      *(v4u*)&sK[key * 72 + c] =
          *(const v4u*)&qkv[((size_t)(b * S_LEN + kb + key)) * (3 * DM) + DM +
                            h * HD + c];
    }
    // Stage V transposed [dim][key]: b32 loads, b16 transposed stores.
#pragma unroll
    for (int i = 0; i < 8; ++i) {
      int e = tid + i * 128;
      int key = e >> 5, d2 = (e & 31) * 2;
      unsigned u = *(const unsigned*)&qkv[((size_t)(b * S_LEN + kb + key)) *
                                              (3 * DM) + 2 * DM + h * HD + d2];
      sVt[d2 * LDS_S + key]       = (unsigned short)u;
      sVt[(d2 + 1) * LDS_S + key] = (unsigned short)(u >> 16);
    }
    __syncthreads();

    if (kb <= qw + 15) {  // wave-uniform causal skip
      // ---- S = Q K^T over 32 keys (2 N-tiles), head-dim 64 (2 K-steps) ----
      v8f sc[2];
#pragma unroll
      for (int j = 0; j < 2; ++j)
#pragma unroll
        for (int r = 0; r < 8; ++r) sc[j][r] = 0.0f;
#pragma unroll
      for (int j = 0; j < 2; ++j)
#pragma unroll
        for (int t = 0; t < 2; ++t) {
          v8u kf = ld_frag(&sK[(j * 16 + l15) * 72 + t * 32 + half * 8], 16);
          sc[j] = wmma_bf16(qf[t], kf, sc[j]);
        }

      // ---- scale + ALiBi + causal mask + online softmax ----
      const int   k0i = kb + l15;
      const int   k1i = k0i + 16;
      const float kb0 = slope * (float)k0i;  // per-lane, once per block
      const float kb1 = slope * (float)k1i;
#pragma unroll
      for (int r = 0; r < 8; ++r) {
        float s0 = fmaf(sc[0][r], 0.125f, kb0 - qb[r]);
        float s1 = fmaf(sc[1][r], 0.125f, kb1 - qb[r]);
        if (k0i > qi[r]) s0 = -__builtin_inff();
        if (k1i > qi[r]) s1 = -__builtin_inff();
        float mx = bfly_max(fmaxf(s0, s1));   // reduce over this row's 16 lanes
        float m_new = fmaxf(m_run[r], mx);
        float alpha = __expf(m_run[r] - m_new);
        float e0 = __expf(s0 - m_new);
        float e1 = __expf(s1 - m_new);
        float sum = bfly_sum(e0 + e1);
        l_run[r] = l_run[r] * alpha + sum;
        m_run[r] = m_new;
#pragma unroll
        for (int n = 0; n < 4; ++n) acc[n][r] *= alpha;
        // Stage P (16x32) for C-format -> A-format re-layout.
        sP[wid * 512 + (half * 8 + r) * 32 + l15]      = f2bf(e0);
        sP[wid * 512 + (half * 8 + r) * 32 + 16 + l15] = f2bf(e1);
      }
      // Order wave-private LDS store -> load (DS ops in-order per wave).
      asm volatile("s_wait_dscnt 0" ::: "memory");

      // ---- O += P @ V ----
      v8u pf = ld_frag(&sP[wid * 512 + l15 * 32 + half * 8], 16);
#pragma unroll
      for (int n = 0; n < 4; ++n) {
        v8u vf = ld_frag(&sVt[(n * 16 + l15) * LDS_S + half * 8], 16);
        acc[n] = wmma_bf16(pf, vf, acc[n]);
      }
    }
    __syncthreads();
  }

  // Normalize + write bf16 [B,S,H*HD].
#pragma unroll
  for (int r = 0; r < 8; ++r) {
    float inv = 1.0f / l_run[r];
#pragma unroll
    for (int n = 0; n < 4; ++n) {
      out[((size_t)(b * S_LEN + qi[r])) * DM + h * HD + n * 16 + l15] =
          f2bf(acc[n][r] * inv);
    }
  }
}

// ---------------------------------------------------------------------------
// Host launcher.
// ---------------------------------------------------------------------------
extern "C" void kernel_launch(void* const* d_in, const int* in_sizes, int n_in,
                              void* d_out, int out_size, void* d_ws, size_t ws_size,
                              hipStream_t stream) {
  (void)in_sizes; (void)n_in; (void)out_size; (void)ws_size;
  const float* x    = (const float*)d_in[0];
  const float* Wqkv = (const float*)d_in[1];
  const float* bqkv = (const float*)d_in[2];
  const float* Wout = (const float*)d_in[3];
  const float* bout = (const float*)d_in[4];

  const int M = 2 * S_LEN;  // 4096 (B*S)
  char* ws = (char*)d_ws;
  unsigned short* xb    = (unsigned short*)ws;                 // 4096x1024
  unsigned short* wqkvb = xb + (size_t)M * DM;                 // 1024x3072
  unsigned short* woutb = wqkvb + (size_t)DM * 3 * DM;         // 1024x1024
  unsigned short* qkvb  = woutb + (size_t)DM * DM;             // 4096x3072
  unsigned short* attnb = qkvb + (size_t)M * 3 * DM;           // 4096x1024
  float* outp = (float*)d_out;

  cvt_f32_bf16<<<(M * DM) / 8 / 256, 256, 0, stream>>>(x, xb, (M * DM) / 8);
  cvt_f32_bf16<<<(DM * 3 * DM) / 8 / 256, 256, 0, stream>>>(
      Wqkv, wqkvb, (DM * 3 * DM) / 8);
  cvt_f32_bf16<<<(DM * DM) / 8 / 256, 256, 0, stream>>>(
      Wout, woutb, (DM * DM) / 8);

  gemm_bf16_wmma<false><<<dim3(3 * DM / 128, M / 128), 256, 0, stream>>>(
      xb, wqkvb, bqkv, qkvb, 3 * DM, DM, DM, 3 * DM, 3 * DM);

  attn_alibi_wmma<<<dim3(S_LEN / 64, NH, 2), 128, 0, stream>>>(qkvb, attnb);

  gemm_bf16_wmma<true><<<dim3(DM / 128, M / 128), 256, 0, stream>>>(
      attnb, woutb, bout, outp, DM, DM, DM, DM, DM);
}
